// TeamActorNet_51960514347495
// MI455X (gfx1250) — matrix operand order repaired and checked by reading, not compiled, not run on hardware
//
#include <hip/hip_runtime.h>

typedef __attribute__((ext_vector_type(16))) __bf16 v16bf;
typedef __attribute__((ext_vector_type(8)))  __bf16 v8bf;
typedef __attribute__((ext_vector_type(8)))  float  v8f;
typedef __attribute__((ext_vector_type(4)))  unsigned int u32x4;

#define BATCH   4096
#define TEAM_N  (BATCH * 256)   // offset of agent_features in d_out

// d_ws layout (bf16 elements): bsw1 [0,18432)  bsw2 [18432,36864)  bsw3 [36864,73728)
// Each conv's weights stored tile-major: tile = kt*NT + nt, 512 bf16 per tile,
// within tile: lane-major (32 lanes x 16 halves), lane l holds
// K = kt*32 + (l>=16?16:0) + h  (h=0..15), N = nt*16 + (l&15).

__global__ __launch_bounds__(256) void tan_swizzle_w(
    const float* __restrict__ w1, const float* __restrict__ w2,
    const float* __restrict__ w3, __bf16* __restrict__ out)
{
  int idx = blockIdx.x * 256 + threadIdx.x;
  if (idx >= 73728) return;
  int rel, NT, conv;
  if (idx < 18432)      { conv = 0; rel = idx;         NT = 2; }
  else if (idx < 36864) { conv = 1; rel = idx - 18432; NT = 4; }
  else                  { conv = 2; rel = idx - 36864; NT = 4; }
  int tile = rel >> 9;
  int r    = rel & 511;
  int lane = r >> 4;
  int h    = r & 15;
  int kt = tile / NT;
  int nt = tile - kt * NT;
  int kg = kt * 32 + ((lane >= 16) ? 16 : 0) + h;   // global K index
  int n  = nt * 16 + (lane & 15);                   // output channel
  float v;
  if (conv == 0)      { int tap = kg >> 6, f = kg & 63; v = w1[(tap * 64 + f) * 32 + n]; }
  else if (conv == 1) { int tap = kg >> 5, c = kg & 31; v = w2[(tap * 32 + c) * 64 + n]; }
  else                { int tap = kg >> 6, c = kg & 63; v = w3[(tap * 64 + c) * 64 + n]; }
  out[idx] = (__bf16)v;
}

__device__ __forceinline__ v8f wmma_bf16(v16bf a, v16bf b, v8f c) {
  // D = A(16x32 bf16) * B(32x16 bf16) + C(16x16 f32)
  return __builtin_amdgcn_wmma_f32_16x16x32_bf16(false, a, false, b, (short)0, c,
                                                 false, false);
}

// A fragment: row of 32 contiguous bf16 in LDS.
// lane<16: halves = K[0..15]; lane>=16: K[16..31]
__device__ __forceinline__ v16bf lda(const __bf16* row, int ln) {
  const int kb = (ln >= 16) ? 16 : 0;
  v8bf lo = *(const v8bf*)(row + kb);
  v8bf hi = *(const v8bf*)(row + kb + 8);
  return __builtin_shufflevector(lo, hi, 0,1,2,3,4,5,6,7,8,9,10,11,12,13,14,15);
}

// B fragment: pre-swizzled tile, lane-contiguous 16 halves
__device__ __forceinline__ v16bf ldb(const __bf16* tile, int ln) {
  v8bf lo = *(const v8bf*)(tile + ln * 16);
  v8bf hi = *(const v8bf*)(tile + ln * 16 + 8);
  return __builtin_shufflevector(lo, hi, 0,1,2,3,4,5,6,7,8,9,10,11,12,13,14,15);
}

// Async global->LDS copy (CDNA5 TDM-lite path, ASYNCcnt tracked).
// lds_off = wave-relative LDS byte offset (low 32 bits of the flat pointer).
__device__ __forceinline__ void async_ld16(unsigned lds_off, const void* gaddr) {
  asm volatile("global_load_async_to_lds_b128 %0, %1, off"
               :: "v"(lds_off), "v"(gaddr) : "memory");
}
__device__ __forceinline__ void wait_async0() {
  asm volatile("s_wait_asynccnt 0x0" ::: "memory");
}

// LDS layout (bytes). Region at 73728 first holds the f32 minimap accumulator
// (64KB) + counts (1KB), then is reused for bsw3 (72KB) after finalize.
#define L_BSW1   0
#define L_BSW2   36864
#define L_REGX   73728
#define L_CNT    139264
#define L_MMBF   147456
#define L_A1     180224
#define L_A2     184320
#define L_O3     186368
#define L_RED    187392
#define L_ZERO   188416
#define L_TOTAL  188480

__global__ __launch_bounds__(256) void tan_fused(
    const float* __restrict__ gf,  const float* __restrict__ myf,
    const float* __restrict__ lid, const float* __restrict__ ef,
    const float* __restrict__ lkef,const float* __restrict__ myp,
    const float* __restrict__ ep,  const float* __restrict__ lkp,
    const float* __restrict__ em,  const float* __restrict__ b1,
    const float* __restrict__ b2,  const float* __restrict__ b3,
    const float* __restrict__ lns, const float* __restrict__ lnb,
    const __bf16* __restrict__ wsw, float* __restrict__ out)
{
  __shared__ __align__(16) unsigned char smem[L_TOTAL];
  __bf16* bsw1 = (__bf16*)(smem + L_BSW1);
  __bf16* bsw2 = (__bf16*)(smem + L_BSW2);
  float*  mmf  = (float* )(smem + L_REGX);
  float*  cnts = (float* )(smem + L_CNT);
  __bf16* bsw3 = (__bf16*)(smem + L_REGX);   // alias, valid after phase 3a
  __bf16* mmbf = (__bf16*)(smem + L_MMBF);
  __bf16* a1s  = (__bf16*)(smem + L_A1);
  __bf16* a2s  = (__bf16*)(smem + L_A2);
  float*  o3s  = (float* )(smem + L_O3);
  float*  red  = (float* )(smem + L_RED);
  const __bf16* zrow = (const __bf16*)(smem + L_ZERO);

  const int t  = threadIdx.x;
  const int wv = t >> 5;
  const int ln = t & 31;
  const int b  = blockIdx.x;
  const unsigned lds_base = (unsigned)(size_t)(void*)smem;

  // ---- Phase 0: async-load conv1+conv2 weights, zero accumulator + zero row ----
  {
    for (int i = t; i < 4608; i += 256)
      async_ld16(lds_base + L_BSW1 + i * 16, (const char*)wsw + i * 16);
    u32x4 z = {0u, 0u, 0u, 0u};
    u32x4* regx = (u32x4*)(smem + L_REGX);
    for (int i = t; i < 4160; i += 256) regx[i] = z;           // mm_f32 + counts
    if (t < 4) ((u32x4*)(smem + L_ZERO))[t] = z;               // zero A-row
    const char* pf = (const char*)wsw + 73728;                 // bsw3 bytes
    for (int off = t * 128; off < 73728; off += 256 * 128)
      __builtin_prefetch(pf + off, 0, 0);                      // global_prefetch_b8
  }
  __syncthreads();

  // ---- Phase 1: scatter 18 entities into f32 minimap via LDS atomics ----
  for (int e = wv; e < 18; e += 8) {
    int g = e / 6, ti = e - g * 6;
    const float *feat, *pp; float mk = 1.0f;
    if (g == 0)      { feat = myf;  pp = myp; }
    else if (g == 1) { feat = ef;   pp = ep;  mk = em[b * 6 + ti]; }
    else             { feat = lkef; pp = lkp; }
    const float* fb = feat + (size_t)(b * 6 + ti) * 64;
    if (fb[0] == 1.0f && mk == 1.0f) {
      float px = pp[(b * 6 + ti) * 2 + 0];
      float py = pp[(b * 6 + ti) * 2 + 1];
      int ix = (int)(px * 16.0f); ix = ix < 0 ? 0 : (ix > 15 ? 15 : ix);
      int iy = (int)(py * 16.0f); iy = iy < 0 ? 0 : (iy > 15 ? 15 : iy);
      float* row = mmf + (iy * 16 + ix) * 64;
      atomicAdd(row + ln, fb[ln]);            // ds_add_f32
      atomicAdd(row + ln + 32, fb[ln + 32]);
      if (ln == 0) atomicAdd(cnts + (iy * 16 + ix), 1.0f);
    }
  }
  __syncthreads();

  // ---- Phase 2: minimap = (gf + sum) / (1 + count)  -> bf16 ----
  {
    int f  = t & 63;
    int cg = t >> 6;
    float gv = gf[b * 64 + f];
    for (int i = 0; i < 64; ++i) {
      int c = cg + i * 4;
      float v = (gv + mmf[c * 64 + f]) / (1.0f + cnts[c]);
      mmbf[c * 64 + f] = (__bf16)v;
    }
  }
  wait_async0();               // bsw1/bsw2 resident before anyone passes this barrier
  __syncthreads();

  // ---- Phase 3a: async-stream conv3 weights into the dead accumulator region ----
  for (int i = t; i < 4608; i += 256)
    async_ld16(lds_base + L_REGX + i * 16, (const char*)wsw + 73728 + i * 16);

  // ---- Phase 3b: conv1  16x16x64 -> 8x8x32, M=64 N=32 K=576, 8 wave-tiles ----
  {
    int mt = wv >> 1, nt = wv & 1;
    int m  = mt * 16 + (ln & 15);
    int oy = m >> 3, ox = m & 7;
    v8f acc = {};
    for (int kt = 0; kt < 18; ++kt) {
      int tap = kt >> 1, fof = (kt & 1) * 32;
      int ty = tap / 3, tx = tap - ty * 3;
      int iy = 2 * oy + ty, ix = 2 * ox + tx;      // SAME: pad_lo=0, pad_hi=1
      const __bf16* src = (iy < 16 && ix < 16)
                        ? (mmbf + (iy * 16 + ix) * 64 + fof) : zrow;
      v16bf a  = lda(src, ln);
      v16bf bb = ldb(bsw1 + (kt * 2 + nt) * 512, ln);
      acc = wmma_bf16(a, bb, acc);
    }
    int n = nt * 16 + (ln & 15);
    float bias = b1[n];
    int m0 = mt * 16 + ((ln >= 16) ? 8 : 0);
#pragma unroll
    for (int r = 0; r < 8; ++r) {
      float v = acc[r] + bias;
      v = v > 0.0f ? v : 0.01f * v;
      a1s[(m0 + r) * 32 + n] = (__bf16)v;
    }
  }
  __syncthreads();

  // ---- Phase 4: conv2 (waves 0-3), agent-feature copy (waves 4-7) ----
  if (wv < 4) {
    int nt = wv;
    int m = ln & 15, oy = m >> 2, ox = m & 3;     // 4x4 output pixels
    v8f acc = {};
    for (int kt = 0; kt < 9; ++kt) {
      int ty = kt / 3, tx = kt - ty * 3;
      int iy = 2 * oy + ty, ix = 2 * ox + tx;
      const __bf16* src = (iy < 8 && ix < 8) ? (a1s + (iy * 8 + ix) * 32) : zrow;
      v16bf a  = lda(src, ln);
      v16bf bb = ldb(bsw2 + (kt * 4 + nt) * 512, ln);
      acc = wmma_bf16(a, bb, acc);
    }
    int n = nt * 16 + (ln & 15);
    float bias = b2[n];
    int m0 = (ln >= 16) ? 8 : 0;
#pragma unroll
    for (int r = 0; r < 8; ++r) {
      float v = acc[r] + bias;
      v = v > 0.0f ? v : 0.01f * v;
      a2s[(m0 + r) * 64 + n] = (__bf16)v;
    }
  } else {
    int tt = t - 128;
    for (int i = tt; i < 576; i += 128) {        // [T=6][96] concat copy
      int e = i / 96, c = i - e * 96;
      float v = (c < 64) ? myf[(b * 6 + e) * 64 + c]
                         : lid[(b * 6 + e) * 32 + (c - 64)];
      out[TEAM_N + (b * 6 + e) * 96 + c] = v;
    }
  }
  wait_async0();               // bsw3 resident before anyone passes this barrier
  __syncthreads();

  // ---- Phase 5: conv3  4x4x64 -> 2x2x64, M=4 (padded to 16) N=64 K=576 ----
  if (wv < 4) {
    int nt = wv;
    int m = ln & 15, oy = m >> 1, ox = m & 1;
    v8f acc = {};
    for (int kt = 0; kt < 18; ++kt) {
      int tap = kt >> 1, fof = (kt & 1) * 32;
      int ty = tap / 3, tx = tap - ty * 3;
      int iy = 2 * oy + ty, ix = 2 * ox + tx;
      const __bf16* src = (m < 4 && iy < 4 && ix < 4)
                        ? (a2s + (iy * 4 + ix) * 64 + fof) : zrow;
      v16bf a  = lda(src, ln);
      v16bf bb = ldb(bsw3 + (kt * 4 + nt) * 512, ln);
      acc = wmma_bf16(a, bb, acc);
    }
    if (ln < 16) {                                // only D rows 0..3 are real
      int n = nt * 16 + ln;
      float bias = b3[n];
#pragma unroll
      for (int r = 0; r < 4; ++r) o3s[r * 64 + n] = acc[r] + bias;
    }
  }
  __syncthreads();

  // ---- Phase 6: LayerNorm(256) + leaky -> team_features ----
  {
    float x = o3s[t];
    red[t] = x;
    __syncthreads();
    for (int s = 128; s > 0; s >>= 1) {
      if (t < s) red[t] += red[t + s];
      __syncthreads();
    }
    float mu = red[0] * (1.0f / 256.0f);
    __syncthreads();
    float d = x - mu;
    red[t] = d * d;
    __syncthreads();
    for (int s = 128; s > 0; s >>= 1) {
      if (t < s) red[t] += red[t + s];
      __syncthreads();
    }
    float var = red[0] * (1.0f / 256.0f);
    float y = d * rsqrtf(var + 1e-6f) * lns[t] + lnb[t];
    y = y > 0.0f ? y : 0.01f * y;
    out[b * 256 + t] = y;
  }
}

extern "C" void kernel_launch(void* const* d_in, const int* in_sizes, int n_in,
                              void* d_out, int out_size, void* d_ws, size_t ws_size,
                              hipStream_t stream) {
  (void)in_sizes; (void)n_in; (void)out_size; (void)ws_size;
  const float* gf   = (const float*)d_in[0];
  const float* myf  = (const float*)d_in[1];
  const float* lid  = (const float*)d_in[2];
  const float* ef   = (const float*)d_in[3];
  const float* lkef = (const float*)d_in[4];
  const float* myp  = (const float*)d_in[5];
  const float* ep   = (const float*)d_in[6];
  const float* lkp  = (const float*)d_in[7];
  const float* em   = (const float*)d_in[8];
  const float* w1   = (const float*)d_in[9];
  const float* b1   = (const float*)d_in[10];
  const float* w2   = (const float*)d_in[11];
  const float* b2   = (const float*)d_in[12];
  const float* w3   = (const float*)d_in[13];
  const float* b3   = (const float*)d_in[14];
  const float* lns  = (const float*)d_in[15];
  const float* lnb  = (const float*)d_in[16];
  __bf16* wsw = (__bf16*)d_ws;
  float* out = (float*)d_out;

  tan_swizzle_w<<<288, 256, 0, stream>>>(w1, w2, w3, wsw);
  tan_fused<<<BATCH, 256, 0, stream>>>(gf, myf, lid, ef, lkef, myp, ep, lkp, em,
                                       b1, b2, b3, lns, lnb, wsw, out);
}
